// GNNLocalCluster0_6158983102547
// MI455X (gfx1250) — matrix operand order repaired and checked by reading, not compiled, not run on hardware
//
#include <hip/hip_runtime.h>
#include <hip/hip_bf16.h>

typedef __attribute__((ext_vector_type(16))) _Float16     v16h;
typedef __attribute__((ext_vector_type(8)))  float        v8f;
typedef __attribute__((ext_vector_type(4)))  unsigned int u32x4;
typedef __attribute__((ext_vector_type(8)))  unsigned int u32x8;

#define TILES   784      // B * 7 * 7
#define N_NODES 256      // 16 x 16 nodes per tile
#define CF      16
#define CIN     256
#define HW      12544    // 112*112
#define WIMG    112
#define KNN     9

static __device__ __forceinline__ v8f wmma_f16(v16h a, v16h b, v8f c) {
  // D = A(16x32 f16) * B(32x16 f16) + C(16x16 f32)
  return __builtin_amdgcn_wmma_f32_16x16x32_f16(false, a, false, b, (short)0, c,
                                                false, false);
}

// ---------------------------------------------------------------------------
// TDM descriptor helpers (CDNA5 Tensor Data Mover, CDNA5 ISA
// §8). Group0: count=1 | lds_addr | global_addr | type=2. Group1: data_size,
// tensor dims/strides, tile dims. 3D form uses groups 2+3; 1D/2D form uses
// the 2-operand encoding. Issued by one wave (TDM ignores EXEC), tracked by
// TENSORcnt.
// ---------------------------------------------------------------------------
static __device__ __forceinline__ void tdm_load_3d(unsigned long long gaddr,
                                                   unsigned int lds_off) {
  // tile: 16 contiguous f32 (X) x 16 rows stride 112 (Y) x 32 ch stride 12544 (Z)
  u32x4 g0;
  g0[0] = 1u;                                            // count=1
  g0[1] = lds_off;                                       // lds_addr (bytes)
  g0[2] = (unsigned int)(gaddr & 0xFFFFFFFFu);           // global_addr[31:0]
  g0[3] = (unsigned int)((gaddr >> 32) & 0x1FFFFFFu) | (2u << 30); // type=2
  u32x8 g1;
  g1[0] = 2u << 16;                                      // data_size=4B
  g1[1] = (112u & 0xFFFFu) << 16;                        // tensor_dim0=112
  g1[2] = 0u | ((112u & 0xFFFFu) << 16);                 // tensor_dim1=112
  g1[3] = 0u | (16u << 16);                              // tile_dim0=16
  g1[4] = 16u | (32u << 16);                             // tile_dim1=16 tile_dim2=32
  g1[5] = 112u;                                          // dim0_stride=112
  g1[6] = 0u | ((12544u & 0xFFFFu) << 16);               // dim1_stride=12544
  g1[7] = 0u;
  u32x4 g2;
  g2[0] = 256u;                                          // tensor_dim2=256
  g2[1] = 0u; g2[2] = 0u; g2[3] = 0u;                    // no dim3, no iterate
  u32x4 g3;
  g3[0] = 0u; g3[1] = 0u; g3[2] = 0u; g3[3] = 0u;
  asm volatile("tensor_load_to_lds %0, %1, %2, %3"
               :: "s"(g0), "s"(g1), "s"(g2), "s"(g3) : "memory");
}

static __device__ __forceinline__ void tdm_load_1d(unsigned long long gaddr,
                                                   unsigned int lds_off,
                                                   unsigned int nelem) {
  u32x4 g0;
  g0[0] = 1u;
  g0[1] = lds_off;
  g0[2] = (unsigned int)(gaddr & 0xFFFFFFFFu);
  g0[3] = (unsigned int)((gaddr >> 32) & 0x1FFFFFFu) | (2u << 30);
  u32x8 g1;
  g1[0] = 2u << 16;                                      // data_size=4B
  g1[1] = (nelem & 0xFFFFu) << 16;                       // tensor_dim0
  g1[2] = ((nelem >> 16) & 0xFFFFu) | (1u << 16);        // tensor_dim1=1
  g1[3] = 0u | ((nelem & 0xFFFFu) << 16);                // tile_dim0
  g1[4] = 0u;                                            // tile_dim1/2 unused
  g1[5] = nelem;                                         // dim0_stride
  g1[6] = 0u;
  g1[7] = 0u;
  asm volatile("tensor_load_to_lds %0, %1" :: "s"(g0), "s"(g1) : "memory");
}

// ---------------------------------------------------------------------------
// Stage 1: f = f_w @ x per tile -> nodes[tile, n, cf].  Streams the 205MB
// input exactly once via TDM into a 2x32KB LDS double buffer; WMMA consumes
// chunk k while the TDM streams chunk k+1 (TENSORcnt overlap).
// ---------------------------------------------------------------------------
__global__ void __launch_bounds__(256) k_fconv(const float* __restrict__ x,
                                               const float* __restrict__ fw,
                                               const float* __restrict__ fb,
                                               float* __restrict__ nodes) {
  __shared__ float s_x[2][32 * 256];     // 64KB: [c_local][wi*16+hj]
  __shared__ float s_fw[CF * CIN];       // 16KB

  const int tile = blockIdx.x;
  const int b  = tile / 49;
  const int r  = tile % 49;
  const int gi = r / 7, gj = r % 7;
  const int tid = threadIdx.x;
  const int wv = tid >> 5;
  const int l  = tid & 31;               // wave32 lane
  const int m  = l & 15;
  const int hi = (l >> 4) & 1;
  const int kbA = hi * 8;
  const int kbB = hi * 16;

  for (int i = tid; i < CF * CIN; i += 256) s_fw[i] = fw[i];

  // global element offset of this tile's corner for channel 0
  const size_t corner = (size_t)b * CIN * HW + (size_t)(gi * 16) * WIMG + gj * 16;
  const unsigned long long xbase = (unsigned long long)(uintptr_t)x;

  if (tid < 32) {                        // wave 0 drives the TDM
    tdm_load_3d(xbase + (corner + (size_t)0 * HW) * 4ull,
                (unsigned int)(uintptr_t)&s_x[0][0]);
    __builtin_amdgcn_s_wait_tensorcnt(0);
  }
  __syncthreads();

  v8f acc0 = {}, acc1 = {};
  for (int ch = 0; ch < 8; ++ch) {       // K chunks of 32 channels
    const int buf = ch & 1;
    if (ch + 1 < 8 && tid < 32)
      tdm_load_3d(xbase + (corner + (size_t)(ch + 1) * 32 * HW) * 4ull,
                  (unsigned int)(uintptr_t)&s_x[buf ^ 1][0]);

    // B operand: f_w^T chunk, K = i + (lane<16?0:16)
    v16h B;
#pragma unroll
    for (int i = 0; i < 16; ++i)
      B[i] = (_Float16)s_fw[m * CIN + ch * 32 + i + kbB];

    // A operands from LDS chunk: K = i + (lane<16?0:16)
    v16h A0, A1;
#pragma unroll
    for (int i = 0; i < 16; ++i) {
      const int cl = i + kbB;          // local channel 0..31
      A0[i] = (_Float16)s_x[buf][cl * 256 + (wv    ) * 16 + m];
      A1[i] = (_Float16)s_x[buf][cl * 256 + (wv + 8) * 16 + m];
    }
    acc0 = wmma_f16(A0, B, acc0);
    acc1 = wmma_f16(A1, B, acc1);

    if (ch + 1 < 8 && tid < 32) __builtin_amdgcn_s_wait_tensorcnt(0);
    __syncthreads();
  }

  const float bias = fb[m];              // cf == D column == lane&15
#pragma unroll
  for (int nbi = 0; nbi < 2; ++nbi) {
    const int nb = wv + nbi * 8;
    const v8f acc = nbi ? acc1 : acc0;
    const size_t base = ((size_t)tile * N_NODES + nb * 16) * CF + m;
#pragma unroll
    for (int rr = 0; rr < 8; ++rr) {
      const int M = rr + hi * 8;
      nodes[base + (size_t)M * CF] = acc[rr] + bias;
    }
  }
}

// ---------------------------------------------------------------------------
// Stage 2: per-tile cosine sim (WMMA, K=16 zero-padded), top-9, sigmoid
// edge weights, weighted gather. One tile per block, all data in LDS
// (tile nodes pulled in with a 1D TDM load).
// ---------------------------------------------------------------------------
__global__ void __launch_bounds__(256) k_graph(const float* __restrict__ nodes,
                                               const float* __restrict__ ea,
                                               const float* __restrict__ eb,
                                               float* __restrict__ mid) {
  __shared__ float s_nodes[N_NODES * CF];   // 16 KB
  __shared__ float s_nrm[N_NODES];
  __shared__ float s_blk[8][16][17];        // padded vs bank conflicts
  __shared__ float s_tw[8][16][KNN];
  __shared__ int   s_ti[8][16][KNN];

  const int tile = blockIdx.x;
  const int tid = threadIdx.x;
  const int wv = tid >> 5;
  const int l  = tid & 31;
  const int m  = l & 15;
  const int hi = (l >> 4) & 1;
  const float alpha = ea[0], beta = eb[0];

  const float* np = nodes + (size_t)tile * N_NODES * CF;
  if (tid < 32) {
    tdm_load_1d((unsigned long long)(uintptr_t)np,
                (unsigned int)(uintptr_t)&s_nodes[0], N_NODES * CF);
    __builtin_amdgcn_s_wait_tensorcnt(0);
  }
  __syncthreads();
  {
    float s = 0.f;
#pragma unroll
    for (int c = 0; c < CF; ++c) { float v = s_nodes[tid * CF + c]; s += v * v; }
    s_nrm[tid] = sqrtf(s);
  }
  __syncthreads();

  for (int nbi = 0; nbi < 2; ++nbi) {
    const int ni = wv + nbi * 8;
    // A: rows n = ni*16+m, halves 0..7 hold cf = hi*8 + i, rest zero (K pad)
    v16h A;
#pragma unroll
    for (int i = 0; i < 8; ++i)
      A[i] = (_Float16)s_nodes[(ni * 16 + m) * CF + i + hi * 8];
#pragma unroll
    for (int i = 8; i < 16; ++i) A[i] = (_Float16)0.f;

    float tv[KNN]; int ti[KNN];
#pragma unroll
    for (int k = 0; k < KNN; ++k) { tv[k] = -1e30f; ti[k] = 0; }

    for (int nj = 0; nj < 16; ++nj) {
      // B: halves 0..7 hold cf = hi*8 + i of col node, rest zero (K pad)
      v16h B;
#pragma unroll
      for (int i = 0; i < 16; ++i)
        B[i] = (i < 8) ? (_Float16)s_nodes[(nj * 16 + m) * CF + i + hi * 8]
                         : (_Float16)0.f;
      v8f d = {};
      d = wmma_f16(A, B, d);
#pragma unroll
      for (int rr = 0; rr < 8; ++rr) {
        const int M = rr + hi * 8;
        const int row = ni * 16 + M;
        const int col = nj * 16 + m;
        const float den = fmaxf(s_nrm[row] * s_nrm[col], 1e-8f);
        s_blk[wv][M][m] = d[rr] / den;
      }
      __syncthreads();
      if (l < 16) {                       // lane owns one row: top-9 insert
#pragma unroll
        for (int c = 0; c < 16; ++c) {
          const float v = s_blk[wv][l][c];
          const int j = nj * 16 + c;
          if (v > tv[KNN - 1]) {
            int k = KNN - 1;
            while (k > 0 && v > tv[k - 1]) {
              tv[k] = tv[k - 1]; ti[k] = ti[k - 1]; --k;
            }
            tv[k] = v; ti[k] = j;
          }
        }
      }
      __syncthreads();
    }

    if (l < 16) {
      float we[KNN]; float s = 0.f;
#pragma unroll
      for (int k = 0; k < KNN; ++k) {
        we[k] = 1.f / (1.f + __expf(-(beta + alpha * tv[k])));
        s += we[k];
      }
      s += 1e-12f;
#pragma unroll
      for (int k = 0; k < KNN; ++k) {
        s_tw[wv][l][k] = we[k] / s;
        s_ti[wv][l][k] = ti[k];
      }
    }
    __syncthreads();
    // weighted gather: 16 rows x 16 cf = 256 items over 32 lanes
#pragma unroll
    for (int t = 0; t < 8; ++t) {
      const int item = l + 32 * t;
      const int row = item >> 4;
      const int cf  = item & 15;
      float acc = 0.f;
#pragma unroll
      for (int k = 0; k < KNN; ++k)
        acc += s_tw[wv][row][k] * s_nodes[s_ti[wv][row][k] * CF + cf];
      mid[((size_t)tile * N_NODES + ni * 16 + row) * CF + cf] = acc;
    }
    __syncthreads();
  }
}

// ---------------------------------------------------------------------------
// Stage 3: out = p_w @ mid (+p_b), inverse tile rearrange fused into the
// address math. Wave handles one 16-pixel group x all 256 out-channels:
// 16 WMMAs of (16 px x 16 cf) * (16 cf x 16 co).  Streams 205MB out once.
// ---------------------------------------------------------------------------
__global__ void __launch_bounds__(256) k_pconv(const float* __restrict__ mid,
                                               const float* __restrict__ pw,
                                               const float* __restrict__ pb,
                                               float* __restrict__ out) {
  const int tid = threadIdx.x;
  const int wv = tid >> 5;
  const int l  = tid & 31;
  const int m  = l & 15;
  const int hi = (l >> 4) & 1;
  const int g = blockIdx.x * 8 + wv;     // 0..12543 pixel groups
  const int b = g / 784;
  const int rem = g % 784;
  const int y = rem / 7;
  const int q = rem % 7;                 // x-tile
  const int gi = y >> 4, wi = y & 15;
  const int tile = (b * 7 + gi) * 7 + q;
  const int x0 = q * 16;

  // A: 16 pixels (rows) x cf (K, padded to 32); 8 contiguous floats per lane
  const float* ap = mid + ((size_t)tile * N_NODES + wi * 16 + m) * CF;
  v16h A;
#pragma unroll
  for (int i = 0; i < 8; ++i) A[i] = (_Float16)ap[i + hi * 8];
#pragma unroll
  for (int i = 8; i < 16; ++i) A[i] = (_Float16)0.f;

  for (int cob = 0; cob < 16; ++cob) {
    v16h B;                              // cf x co block of p_w
#pragma unroll
    for (int i = 0; i < 16; ++i)
      B[i] = (i < 8) ? (_Float16)pw[(cob * 16 + m) * CF + i + hi * 8]
                       : (_Float16)0.f;
    v8f d = {};
    d = wmma_f16(A, B, d);
    const int co = cob * 16 + m;
    const float bias = pb[co];
#pragma unroll
    for (int rr = 0; rr < 8; ++rr) {
      const int M = rr + hi * 8;         // pixel within group
      out[(size_t)(b * CIN + co) * HW + (size_t)y * WIMG + x0 + M] = d[rr] + bias;
    }
  }
}

extern "C" void kernel_launch(void* const* d_in, const int* in_sizes, int n_in,
                              void* d_out, int out_size, void* d_ws, size_t ws_size,
                              hipStream_t stream) {
  const float* x  = (const float*)d_in[0];   // [16,256,112,112]
  const float* fw = (const float*)d_in[1];   // [16,256]
  const float* fb = (const float*)d_in[2];   // [16]
  const float* pw = (const float*)d_in[3];   // [256,16]
  const float* pb = (const float*)d_in[4];   // [256]
  const float* ea = (const float*)d_in[5];   // [1]
  const float* eb = (const float*)d_in[6];   // [1]
  float* out = (float*)d_out;                // [16,256,12544]

  float* nodes = (float*)d_ws;                               // 784*256*16 f32
  float* mid   = nodes + (size_t)TILES * N_NODES * CF;       // same size

  k_fconv<<<TILES, 256, 0, stream>>>(x, fw, fb, nodes);
  k_graph<<<TILES, 256, 0, stream>>>(nodes, ea, eb, mid);
  k_pconv<<<1568, 256, 0, stream>>>(mid, pw, pb, out);
}
